// BidirectionalMamba_88158498718099
// MI455X (gfx1250) — matrix operand order repaired and checked by reading, not compiled, not run on hardware
//
#include <hip/hip_runtime.h>
#include <hip/hip_bf16.h>
#include <cstddef>

// Problem constants (match reference)
#define BATCH   4
#define SEQ_T   4096
#define DMODEL  256
#define DINNER  512
#define DSTATE  16
#define DTRANK  16
#define DCONV   4
#define ROWS    (BATCH * SEQ_T)   // 16384, divisible by 128

// GEMM tiling
#define TILE_M  128
#define TILE_N  64
#define TILE_K  32

typedef __attribute__((ext_vector_type(16))) __bf16 bf16x16;
typedef __attribute__((ext_vector_type(8)))  float  f32x8;
typedef __attribute__((ext_vector_type(4)))  unsigned int u32x4;

union Frag { bf16x16 v; u32x4 q[2]; };

__device__ __forceinline__ unsigned short f2bf(float f) {
  union { float f; unsigned u; } v; v.f = f;
  unsigned u = v.u;
  unsigned r = (u + 0x7FFFu + ((u >> 16) & 1u)) >> 16;  // RNE
  return (unsigned short)r;
}
__device__ __forceinline__ float bf2f(unsigned short h) {
  union { unsigned u; float f; } v; v.u = ((unsigned)h) << 16; return v.f;
}

// CDNA5 async global->LDS copy (16 bytes per lane), tracked by ASYNCcnt.
__device__ __forceinline__ void async_copy_b128(const void* gaddr, void* lds_ptr) {
  unsigned ldsOff = (unsigned)(size_t)lds_ptr;   // LDS aperture: low 32 bits = byte offset
  asm volatile("global_load_async_to_lds_b128 %0, %1, off"
               :: "v"(ldsOff), "v"(gaddr)
               : "memory");
}
__device__ __forceinline__ void wait_async_le3() {
  asm volatile("s_wait_asynccnt 0x3" ::: "memory");
}
__device__ __forceinline__ void wait_async_0() {
  asm volatile("s_wait_asynccnt 0x0" ::: "memory");
}

// ---------------- conversion / transpose prep kernels ----------------

__global__ void k_f2bf(const float* __restrict__ in, unsigned short* __restrict__ out, int n) {
  int i = blockIdx.x * blockDim.x + threadIdx.x;
  if (i < n) out[i] = f2bf(in[i]);
}

// in: (K,N) fp32 row-major  ->  out: (N,K) bf16 row-major (i.e. B transposed)
__global__ void k_wtrans(const float* __restrict__ in, unsigned short* __restrict__ out,
                         int K, int N, int n) {
  int i = blockIdx.x * blockDim.x + threadIdx.x;
  if (i >= n) return;
  int k = i % K, nn = i / K;
  out[i] = f2bf(in[(size_t)k * N + nn]);
}

// ---------------- bf16 WMMA GEMM with async-LDS double buffering ----------------
// C(MxN) = A(MxK) * B(KxN).  A: bf16 row-major.  Bt: bf16 (NxK) row-major.
// M % 128 == 0, K % 32 == 0. Block = 256 thr (8 waves); block tile 128x64;
// wave tile 64(M) x 16(N). A/B K-slices staged in LDS via global_load_async_to_lds_b128,
// double buffered, overlapped with WMMA compute.
template <bool BF16_OUT>
__global__ __launch_bounds__(256) void k_gemm(const unsigned short* __restrict__ A,
                                              const unsigned short* __restrict__ Bt,
                                              void* __restrict__ Cout,
                                              int M, int N, int K) {
  __shared__ unsigned short aLds[2][TILE_M * TILE_K];  // 8 KB per buffer
  __shared__ unsigned short bLds[2][TILE_N * TILE_K];  // 4 KB per buffer

  const int tid   = threadIdx.x;
  const int lane  = tid & 31;
  const int wave  = tid >> 5;
  const int wm    = wave >> 2;          // 0..1  (M strip)
  const int wn    = wave & 3;           // 0..3  (N tile)
  const int lrow  = lane & 15;
  const int lhalf = lane >> 4;

  const int blockM0 = blockIdx.y * TILE_M;
  const int blockN0 = blockIdx.x * TILE_N;
  const int tileN   = blockN0 + wn * 16;
  const int nIdx    = tileN + lrow;
  const bool nOK    = nIdx < N;

  f32x8 acc[4];
  for (int i = 0; i < 4; ++i)
    for (int r = 0; r < 8; ++r) acc[i][r] = 0.0f;

  // Cooperative async stage of one K-slice: A = TILE_M x 32, B = TILE_N x 32.
  // 512 A-chunks (2/lane) + 256 B-chunks (1/lane), 16B each -> 3 async issues per lane.
  auto issue_tile = [&](int buf, int k0) {
#pragma unroll
    for (int rep = 0; rep < 2; ++rep) {
      int c = tid + rep * 256;                 // 0..511
      int row = c >> 2, part = c & 3;
      async_copy_b128(A + (size_t)(blockM0 + row) * K + k0 + part * 8,
                      &aLds[buf][c * 8]);
    }
    {
      int c = tid;                              // 0..255
      int row = c >> 2, part = c & 3;
      async_copy_b128(Bt + (size_t)(blockN0 + row) * K + k0 + part * 8,
                      &bLds[buf][c * 8]);
    }
  };

  issue_tile(0, 0);   // prologue

  int cur = 0;
  for (int k0 = 0; k0 < K; k0 += TILE_K, cur ^= 1) {
    const bool hasNext = (k0 + TILE_K) < K;
    if (hasNext) {
      issue_tile(cur ^ 1, k0 + TILE_K);   // overlap next DMA with this tile's compute
      wait_async_le3();                   // oldest 3 (current buffer) complete, in order
    } else {
      wait_async_0();
    }
    __syncthreads();                      // all waves' portions of buffer `cur` visible

    // B fragment: lane holds K = lhalf*16 .. +15 of column nIdx (contiguous in LDS row)
    Frag bf;
    {
      const unsigned short* br = &bLds[cur][(wn * 16 + lrow) * TILE_K + lhalf * 16];
      bf.q[0] = *reinterpret_cast<const u32x4*>(br);
      bf.q[1] = *reinterpret_cast<const u32x4*>(br + 8);
    }
#pragma unroll
    for (int i = 0; i < 4; ++i) {
      // A fragment: lane holds k [lhalf*16,+16) contiguous
      const unsigned short* ar = &aLds[cur][(wm * 64 + i * 16 + lrow) * TILE_K + lhalf * 16];
      Frag af;
      af.q[0] = *reinterpret_cast<const u32x4*>(ar);
      af.q[1] = *reinterpret_cast<const u32x4*>(ar + 8);
      acc[i] = __builtin_amdgcn_wmma_f32_16x16x32_bf16(
          false, af.v, false, bf.v, (short)0, acc[i], false, false);
    }
    __syncthreads();                      // reads of `cur` done before it is overwritten
  }

  if (!nOK) return;
#pragma unroll
  for (int i = 0; i < 4; ++i) {
    const int rowBase = blockM0 + wm * 64 + i * 16 + lhalf * 8;
    if (BF16_OUT) {
      unsigned short* C = (unsigned short*)Cout;
      for (int r = 0; r < 8; ++r) C[(size_t)(rowBase + r) * N + nIdx] = f2bf(acc[i][r]);
    } else {
      float* C = (float*)Cout;
      for (int r = 0; r < 8; ++r) C[(size_t)(rowBase + r) * N + nIdx] = acc[i][r];
    }
  }
}

// ---------------- depthwise causal conv (window 4) + bias + SiLU ----------------
// G: (B*T, 1024) bf16 in ORIGINAL time; xc = cols [0,512). Output xcout in direction-local time.
__global__ void k_conv_silu(const unsigned short* __restrict__ G,
                            const float* __restrict__ cw, const float* __restrict__ cb,
                            unsigned short* __restrict__ xcout, int dir, int n) {
  int idx = blockIdx.x * blockDim.x + threadIdx.x;
  if (idx >= n) return;
  int d = idx & (DINNER - 1);
  int t = (idx >> 9) & (SEQ_T - 1);
  int b = idx >> 21;
  float acc = cb[d];
#pragma unroll
  for (int j = 0; j < DCONV; ++j) {
    int s = t - (DCONV - 1) + j;
    if (s >= 0) {
      int tor = dir ? (SEQ_T - 1 - s) : s;
      acc += cw[d * DCONV + j] * bf2f(G[((size_t)b * SEQ_T + tor) * (2 * DINNER) + d]);
    }
  }
  acc = acc / (1.f + __expf(-acc));  // SiLU
  xcout[idx] = f2bf(acc);
}

// ---------------- selective scan (sequential over T) ----------------
// One thread per (b, d); 16-state recurrence in registers; dbc row staged in LDS.
__global__ __launch_bounds__(128) void k_scan(const float* __restrict__ dbc,
                                              const unsigned short* __restrict__ xc,
                                              const unsigned short* __restrict__ G,
                                              const float* __restrict__ dtw,
                                              const float* __restrict__ dtb,
                                              const float* __restrict__ Alog,
                                              const float* __restrict__ Dp,
                                              unsigned short* __restrict__ yg, int dir) {
  const int d = blockIdx.x * 128 + threadIdx.x;
  const int b = blockIdx.y;
  __shared__ float sh[DTRANK + 2 * DSTATE];  // 48 floats: dt | B | C

  float w[DTRANK], Av[DSTATE], h[DSTATE];
  for (int r = 0; r < DTRANK; ++r) w[r] = dtw[r * DINNER + d];
  for (int s = 0; s < DSTATE; ++s) { Av[s] = -__expf(Alog[d * DSTATE + s]); h[s] = 0.f; }
  const float bias = dtb[d], Dd = Dp[d];

  for (int tl = 0; tl < SEQ_T; ++tl) {
    __syncthreads();
    if (threadIdx.x < 48) sh[threadIdx.x] = dbc[((size_t)b * SEQ_T + tl) * 48 + threadIdx.x];
    __syncthreads();

    float dtv = bias;
    for (int r = 0; r < DTRANK; ++r) dtv += sh[r] * w[r];
    dtv = (dtv > 20.f) ? dtv : log1pf(__expf(dtv));   // softplus

    const int tor = dir ? (SEQ_T - 1 - tl) : tl;
    const float u  = bf2f(xc[((size_t)b * SEQ_T + tl) * DINNER + d]);
    const float zv = bf2f(G[((size_t)b * SEQ_T + tor) * (2 * DINNER) + DINNER + d]);
    const float du = dtv * u;

    float y = 0.f;
#pragma unroll
    for (int s = 0; s < DSTATE; ++s) {
      float dA = __expf(dtv * Av[s]);
      h[s] = dA * h[s] + du * sh[DTRANK + s];
      y += h[s] * sh[DTRANK + DSTATE + s];
    }
    y += u * Dd;
    y *= zv / (1.f + __expf(-zv));  // * SiLU(z)
    yg[((size_t)b * SEQ_T + tl) * DINNER + d] = f2bf(y);
  }
}

// ---------------- un-flip + concat into (B*T, 512) bf16 ----------------
__global__ void k_concat(const float* __restrict__ ydir, unsigned short* __restrict__ cc,
                         int dir, int n) {
  int idx = blockIdx.x * blockDim.x + threadIdx.x;
  if (idx >= n) return;
  int c = idx & (DMODEL - 1);
  int t = (idx >> 8) & (SEQ_T - 1);
  int b = idx >> 20;
  int tor = dir ? (SEQ_T - 1 - t) : t;
  cc[((size_t)b * SEQ_T + tor) * (2 * DMODEL) + dir * DMODEL + c] = f2bf(ydir[idx]);
}

// ---------------- bias + LayerNorm + exact GELU ----------------
__global__ __launch_bounds__(256) void k_ln_gelu(const float* __restrict__ h,
                                                 const float* __restrict__ bias,
                                                 const float* __restrict__ gamma,
                                                 const float* __restrict__ beta,
                                                 float* __restrict__ out) {
  __shared__ float red[DMODEL];
  __shared__ float s_mu, s_rstd;
  const int row = blockIdx.x, c = threadIdx.x;
  float v = h[(size_t)row * DMODEL + c] + bias[c];
  red[c] = v; __syncthreads();
  for (int off = DMODEL / 2; off > 0; off >>= 1) {
    if (c < off) red[c] += red[c + off];
    __syncthreads();
  }
  if (c == 0) s_mu = red[0] * (1.f / DMODEL);
  __syncthreads();
  const float ctr = v - s_mu;
  red[c] = ctr * ctr; __syncthreads();
  for (int off = DMODEL / 2; off > 0; off >>= 1) {
    if (c < off) red[c] += red[c + off];
    __syncthreads();
  }
  if (c == 0) s_rstd = rsqrtf(red[0] * (1.f / DMODEL) + 1e-5f);
  __syncthreads();
  const float g = ctr * s_rstd * gamma[c] + beta[c];
  out[(size_t)row * DMODEL + c] = 0.5f * g * (1.f + erff(g * 0.70710678118654752f));
}

// ---------------- host side ----------------

static inline dim3 gemm_grid(int M, int N) { return dim3((N + 63) / 64, M / 128); }

extern "C" void kernel_launch(void* const* d_in, const int* in_sizes, int n_in,
                              void* d_out, int out_size, void* d_ws, size_t ws_size,
                              hipStream_t stream) {
  (void)in_sizes; (void)n_in; (void)out_size; (void)ws_size;

  const float* x = (const float*)d_in[0];
  const float* p_in[2]  = {(const float*)d_in[1],  (const float*)d_in[10]};
  const float* p_cw[2]  = {(const float*)d_in[2],  (const float*)d_in[11]};
  const float* p_cb[2]  = {(const float*)d_in[3],  (const float*)d_in[12]};
  const float* p_xp[2]  = {(const float*)d_in[4],  (const float*)d_in[13]};
  const float* p_dtw[2] = {(const float*)d_in[5],  (const float*)d_in[14]};
  const float* p_dtb[2] = {(const float*)d_in[6],  (const float*)d_in[15]};
  const float* p_al[2]  = {(const float*)d_in[7],  (const float*)d_in[16]};
  const float* p_D[2]   = {(const float*)d_in[8],  (const float*)d_in[17]};
  const float* p_op[2]  = {(const float*)d_in[9],  (const float*)d_in[18]};
  const float* fw    = (const float*)d_in[19];
  const float* fb    = (const float*)d_in[20];
  const float* fg    = (const float*)d_in[21];
  const float* fbeta = (const float*)d_in[22];
  float* out = (float*)d_out;

  // Workspace carve-out
  char* p = (char*)d_ws;
  auto alloc = [&](size_t bytes) -> void* {
    void* r = (void*)p; p += (bytes + 255) & ~(size_t)255; return r;
  };
  unsigned short* x_bf   = (unsigned short*)alloc((size_t)ROWS * DMODEL * 2);
  unsigned short* wInT[2], *wXpT[2], *wOpT[2];
  for (int d = 0; d < 2; ++d) {
    wInT[d] = (unsigned short*)alloc((size_t)(2 * DINNER) * DMODEL * 2);  // (1024,256)
    wXpT[d] = (unsigned short*)alloc((size_t)48 * DINNER * 2);            // (48,512)
    wOpT[d] = (unsigned short*)alloc((size_t)DMODEL * DINNER * 2);        // (256,512)
  }
  unsigned short* wFuT  = (unsigned short*)alloc((size_t)DMODEL * (2 * DMODEL) * 2); // (256,512)
  unsigned short* Gbuf  = (unsigned short*)alloc((size_t)ROWS * 2 * DINNER * 2);     // per-dir reused
  unsigned short* xcBf  = (unsigned short*)alloc((size_t)ROWS * DINNER * 2);
  float*          dbc   = (float*)alloc((size_t)ROWS * 48 * 4);
  unsigned short* ygBf  = (unsigned short*)alloc((size_t)ROWS * DINNER * 2);
  float*          ydir  = (float*)alloc((size_t)ROWS * DMODEL * 4);
  unsigned short* ccBf  = (unsigned short*)alloc((size_t)ROWS * 2 * DMODEL * 2);
  float*          hbuf  = (float*)alloc((size_t)ROWS * DMODEL * 4);

  // Prep: activations -> bf16, weights -> bf16 transposed
  {
    int n = ROWS * DMODEL;
    k_f2bf<<<(n + 255) / 256, 256, 0, stream>>>(x, x_bf, n);
  }
  for (int d = 0; d < 2; ++d) {
    int n1 = DMODEL * 2 * DINNER;   // in_proj  K=256 N=1024
    k_wtrans<<<(n1 + 255) / 256, 256, 0, stream>>>(p_in[d], wInT[d], DMODEL, 2 * DINNER, n1);
    int n2 = DINNER * 48;           // x_proj   K=512 N=48
    k_wtrans<<<(n2 + 255) / 256, 256, 0, stream>>>(p_xp[d], wXpT[d], DINNER, 48, n2);
    int n3 = DINNER * DMODEL;       // out_proj K=512 N=256
    k_wtrans<<<(n3 + 255) / 256, 256, 0, stream>>>(p_op[d], wOpT[d], DINNER, DMODEL, n3);
  }
  {
    int n = 2 * DMODEL * DMODEL;    // fusion w K=512 N=256
    k_wtrans<<<(n + 255) / 256, 256, 0, stream>>>(fw, wFuT, 2 * DMODEL, DMODEL, n);
  }

  for (int dir = 0; dir < 2; ++dir) {
    // in_proj: (16384x256) x (256x1024) -> G bf16 (original time)
    k_gemm<true><<<gemm_grid(ROWS, 2 * DINNER), 256, 0, stream>>>(
        x_bf, wInT[dir], (void*)Gbuf, ROWS, 2 * DINNER, DMODEL);
    // causal depthwise conv + SiLU -> xc bf16 (local time)
    {
      int n = ROWS * DINNER;
      k_conv_silu<<<(n + 255) / 256, 256, 0, stream>>>(Gbuf, p_cw[dir], p_cb[dir], xcBf, dir, n);
    }
    // x_proj: (16384x512) x (512x48) -> dbc fp32
    k_gemm<false><<<gemm_grid(ROWS, 48), 256, 0, stream>>>(
        xcBf, wXpT[dir], (void*)dbc, ROWS, 48, DINNER);
    // selective scan -> gated y bf16
    k_scan<<<dim3(DINNER / 128, BATCH), 128, 0, stream>>>(
        dbc, xcBf, Gbuf, p_dtw[dir], p_dtb[dir], p_al[dir], p_D[dir], ygBf, dir);
    // out_proj: (16384x512) x (512x256) -> ydir fp32 (local time)
    k_gemm<false><<<gemm_grid(ROWS, DMODEL), 256, 0, stream>>>(
        ygBf, wOpT[dir], (void*)ydir, ROWS, DMODEL, DINNER);
    // un-flip + concat
    {
      int n = ROWS * DMODEL;
      k_concat<<<(n + 255) / 256, 256, 0, stream>>>(ydir, ccBf, dir, n);
    }
  }

  // fusion GEMM: (16384x512) x (512x256) -> hbuf fp32
  k_gemm<false><<<gemm_grid(ROWS, DMODEL), 256, 0, stream>>>(
      ccBf, wFuT, (void*)hbuf, ROWS, DMODEL, 2 * DMODEL);
  // bias + LayerNorm + GELU -> out
  k_ln_gelu<<<ROWS, DMODEL, 0, stream>>>(hbuf, fb, fg, fbeta, out);
}
